// SGRUNET_83288005804286
// MI455X (gfx1250) — matrix operand order repaired and checked by reading, not compiled, not run on hardware
//
#include <hip/hip_runtime.h>

// Problem constants (from reference)
#define B_   64
#define T_   200
#define IN_  10000     // H*W
#define HID_ 50
#define G3_  150       // 3*HID
#define OUT_ 3
#define NCH_ 313       // ceil(IN/32) K-chunks for the big GEMM

typedef __attribute__((ext_vector_type(16))) __bf16 v16bf;
typedef __attribute__((ext_vector_type(8)))  float  v8f;

// Workspace layout (bytes):
//   [0, 3205120)             : swizzled bf16 Wih fragments  (313*10*32*16 halves)
//   [3205120, 3205120+7.68M) : xproj f32 [T, B, 150]
#define SWIZW_HALVES (NCH_ * 10 * 32 * 16)
#define XP_OFFSET_BYTES (SWIZW_HALVES * 2)   // 3,205,120 (256B aligned)
#define XSTEP_BYTES (32 * T_ * 4)            // x advance per K-chunk  (25600)
#define WSTEP_BYTES (10 * 32 * 16 * 2)       // swizW advance per K-chunk (10240)

#if defined(__has_builtin)
#if __has_builtin(__builtin_amdgcn_global_load_async_to_lds_b32)
#define HAVE_ASYNC_LDS 1
#endif
#endif

#ifdef HAVE_ASYNC_LDS
typedef __attribute__((address_space(1))) int gint_t;   // global (AS1) int*
typedef __attribute__((address_space(3))) int lint_t;   // LDS    (AS3) int*
#if __has_builtin(__builtin_amdgcn_s_wait_asynccnt)
#define WAIT_ASYNC0() __builtin_amdgcn_s_wait_asynccnt(0)
#else
#define WAIT_ASYNC0() asm volatile("s_wait_asynccnt 0x0" ::: "memory")
#endif
#endif

// ===================================================================
// Kernel 0: one-time swizzle of Wih into per-lane WMMA B-fragment order.
// ===================================================================
__global__ __launch_bounds__(256) void swizzle_wih_kernel(
    const float* __restrict__ Wih, __bf16* __restrict__ swizW) {
  const int id = blockIdx.x * 256 + threadIdx.x;   // (kc, kk, n), n fastest
  if (id >= NCH_ * 32 * 160) return;
  const int n  = id % 160;
  const int kk = (id / 160) % 32;
  const int kc = id / (160 * 32);
  const int k  = kc * 32 + kk;
  const int kcl = k < IN_ ? k : IN_ - 1;
  const int ncl = n < G3_ ? n : G3_ - 1;
  float v = Wih[kcl * G3_ + ncl];
  v = (k < IN_ && n < G3_) ? v : 0.f;
  const int nf = n >> 4, col = n & 15;
  const int lane = col + 16 * ((kk >> 3) & 1);
  swizW[((kc * 10 + nf) * 32 + lane) * 16 + ((kk & 7) + ((kk & 16) ? 8 : 0))] = (__bf16)v;
}

// ===================================================================
// Kernel 1: xproj[t,b,:] = x[b,:,t] @ Wih + bih  via WMMA bf16.
// 256 thr / 8 waves; M-tile = 64 rows = 4 batches x 16 t.
// Async path: x staged straight into swizzled LDS via
// GLOBAL_LOAD_ASYNC_TO_LDS_B32 (ASYNCcnt) so the B-fragment LOADcnt
// waits inside the WMMA chain never touch the HBM x stream.
// ===================================================================
__global__ __launch_bounds__(256) void xproj_kernel(
    const float* __restrict__ x, const __bf16* __restrict__ swizW,
    const float* __restrict__ bih, float* __restrict__ xp) {
  const int tid   = threadIdx.x;
  const int lane  = tid & 31;
  const int wave  = tid >> 5;
  const int mt    = wave & 3;   // batch within block
  const int nh    = wave >> 2;  // n-half (5 tiles)
  const int col   = lane & 15;
  const int ttile = blockIdx.x % 13;
  const int b0    = (blockIdx.x / 13) * 4;
  const int t0    = ttile * 16;

  // per-thread staging coordinates (8 elements / chunk)
  int sIdx[8];    // swizzled LDS element index
  int xoff[8];    // byte offset from x for chunk 0
  int xoffL[8];   // absolute byte offset for the clamped last chunk (312)
  #pragma unroll
  for (int j = 0; j < 8; ++j) {
    const int i    = tid + j * 256;
    const int trow = i & 15;
    const int krow = (i >> 4) & 31;
    const int smt  = i >> 9;
    const int lnhi = (krow >> 3) & 1;
    const int h    = (krow & 7) + ((krow & 16) ? 8 : 0);
    sIdx[j] = ((smt * 32) + trow + 16 * lnhi) * 16 + h;
    int t = t0 + trow; t = t < T_ ? t : T_ - 1;   // garbage rows masked at store
    const long long rowb = (long long)(b0 + smt) * IN_;
    xoff[j] = (int)((((rowb + krow) * T_) + t) * 4);
    int kl = (NCH_ - 1) * 32 + krow; kl = kl < IN_ ? kl : IN_ - 1;
    xoffL[j] = (int)((((rowb + kl) * T_) + t) * 4);
  }
  const int woff = nh * 5 * 1024 + lane * 32;   // byte offset into swizW chunk

  v8f acc[5];
  #pragma unroll
  for (int i = 0; i < 5; ++i)
    #pragma unroll
    for (int r = 0; r < 8; ++r) acc[i][r] = 0.f;

  const char* xb = (const char*)x;       // bumped by XSTEP_BYTES per chunk
  const char* wb = (const char*)swizW;   // bumped by WSTEP_BYTES per chunk

#ifdef HAVE_ASYNC_LDS
  // ================= async-to-LDS pipeline =================
  __shared__ __align__(64) float swizAf[2][4 * 32 * 16];  // 16 KB, f32 tiles

  lint_t* ld0[8]; lint_t* ld1[8];
  #pragma unroll
  for (int j = 0; j < 8; ++j) {
    ld0[j] = (lint_t*)(int*)&swizAf[0][sIdx[j]];
    ld1[j] = (lint_t*)(int*)&swizAf[1][sIdx[j]];
  }

  auto issue = [&](const char* base, int buf) {
    #pragma unroll
    for (int j = 0; j < 8; ++j)
      __builtin_amdgcn_global_load_async_to_lds_b32(
          (gint_t*)(int*)(base + xoff[j]), buf ? ld1[j] : ld0[j], 0, 0);
  };
  auto issue_last = [&](int buf) {
    #pragma unroll
    for (int j = 0; j < 8; ++j)
      __builtin_amdgcn_global_load_async_to_lds_b32(
          (gint_t*)(int*)((const char*)x + xoffL[j]), buf ? ld1[j] : ld0[j], 0, 0);
  };
  auto mma_chunk = [&](int pbuf, bool mask_hi) {
    v16bf bfr[5];
    #pragma unroll
    for (int nf = 0; nf < 5; ++nf)
      bfr[nf] = *reinterpret_cast<const v16bf*>(wb + woff + nf * 1024);
    const float* ap = &swizAf[pbuf][(mt * 32 + lane) * 16];
    const float4 q0 = *(const float4*)(ap + 0);
    const float4 q1 = *(const float4*)(ap + 4);
    v16bf af;
    af[0] = (__bf16)q0.x; af[1] = (__bf16)q0.y;
    af[2] = (__bf16)q0.z; af[3] = (__bf16)q0.w;
    af[4] = (__bf16)q1.x; af[5] = (__bf16)q1.y;
    af[6] = (__bf16)q1.z; af[7] = (__bf16)q1.w;
    if (mask_hi) {            // last chunk: k in [16,32) invalid -> h >= 8
      #pragma unroll
      for (int h = 8; h < 16; ++h) af[h] = (__bf16)0.f;
    } else {
      const float4 q2 = *(const float4*)(ap + 8);
      const float4 q3 = *(const float4*)(ap + 12);
      af[8]  = (__bf16)q2.x; af[9]  = (__bf16)q2.y;
      af[10] = (__bf16)q2.z; af[11] = (__bf16)q2.w;
      af[12] = (__bf16)q3.x; af[13] = (__bf16)q3.y;
      af[14] = (__bf16)q3.z; af[15] = (__bf16)q3.w;
    }
    #pragma unroll
    for (int nf = 0; nf < 5; ++nf)
      acc[nf] = __builtin_amdgcn_wmma_f32_16x16x32_bf16(
          false, af, false, bfr[nf], (short)0, acc[nf], false, false);
  };

  issue((const char*)x, 0);                 // chunk 0 in flight
  int p = 0;
  for (int kc = 0; kc < NCH_ - 2; ++kc) {   // chunks 0..310
    WAIT_ASYNC0();                          // my chunk-kc scatters landed
    __syncthreads();                        // everyone's landed
    xb += XSTEP_BYTES;
    issue(xb, p ^ 1);                       // chunk kc+1 (ASYNCcnt, decoupled)
    mma_chunk(p, false);
    wb += WSTEP_BYTES;
    p ^= 1;
  }
  WAIT_ASYNC0();  __syncthreads();          // chunk 311 ready
  issue_last(p ^ 1);                        // chunk 312, clamped addresses
  mma_chunk(p, false);
  wb += WSTEP_BYTES;  p ^= 1;
  WAIT_ASYNC0();  __syncthreads();          // chunk 312 ready
  mma_chunk(p, true);                       // zero-mask h >= 8

#else
  // ================= fallback: register staging (round-3 path) ======
  __shared__ __align__(32) __bf16 swizA[2][4 * 32 * 16];  // 8 KB

  float xr[8];
  #pragma unroll
  for (int j = 0; j < 8; ++j) xr[j] = *(const float*)(xb + xoff[j]);

  auto publish = [&](int pbuf, bool mask_tail) {
    #pragma unroll
    for (int j = 0; j < 8; ++j) {
      float v = xr[j];
      if (mask_tail) {
        const int krow = ((tid + j * 256) >> 4) & 31;
        v = (krow < 16) ? v : 0.f;
      }
      swizA[pbuf][sIdx[j]] = (__bf16)v;
    }
  };
  auto mma_chunk = [&](int pbuf) {
    v16bf bfr[5];
    #pragma unroll
    for (int nf = 0; nf < 5; ++nf)
      bfr[nf] = *reinterpret_cast<const v16bf*>(wb + woff + nf * 1024);
    const v16bf af = *reinterpret_cast<const v16bf*>(
        &swizA[pbuf][(mt * 32 + lane) * 16]);
    #pragma unroll
    for (int nf = 0; nf < 5; ++nf)
      acc[nf] = __builtin_amdgcn_wmma_f32_16x16x32_bf16(
          false, af, false, bfr[nf], (short)0, acc[nf], false, false);
  };

  int p = 0;
  for (int kc = 0; kc < NCH_ - 2; ++kc) {
    publish(p, false);
    __syncthreads();
    mma_chunk(p);
    xb += XSTEP_BYTES;
    #pragma unroll
    for (int j = 0; j < 8; ++j) xr[j] = *(const float*)(xb + xoff[j]);
    wb += WSTEP_BYTES;
    p ^= 1;
  }
  publish(p, false);
  __syncthreads();
  #pragma unroll
  for (int j = 0; j < 8; ++j)
    xr[j] = *(const float*)((const char*)x + xoffL[j]);
  mma_chunk(p);
  wb += WSTEP_BYTES;
  p ^= 1;
  publish(p, true);
  __syncthreads();
  mma_chunk(p);
#endif

  // ---- store: D row = r + 8*(lane>=16), col = lane&15 ----
  const int b = b0 + mt;
  #pragma unroll
  for (int nf = 0; nf < 5; ++nf) {
    const int n = (nh * 5 + nf) * 16 + col;
    #pragma unroll
    for (int r = 0; r < 8; ++r) {
      const int tr = t0 + r + ((lane & 16) ? 8 : 0);
      if (tr < T_ && n < G3_)
        xp[((size_t)tr * B_ + b) * G3_ + n] = acc[nf][r] + bih[n];
    }
  }
}

// ===================================================================
// Kernel 2: persistent single-workgroup sequential scan over T=200.
// ===================================================================
__global__ __launch_bounds__(256) void sgru_scan_kernel(
    const float* __restrict__ xp,   const float* __restrict__ Whh,
    const float* __restrict__ bhh,  const float* __restrict__ thr_sgru,
    const float* __restrict__ fcW,  const float* __restrict__ fcb,
    const float* __restrict__ beta_p, const float* __restrict__ thrL_p,
    float* __restrict__ out) {
  __shared__ __bf16 hgS[B_ * G3_];                          // 19200 B
  __shared__ __align__(32) __bf16 swizMem[2 * 4 * 32 * 16]; //  8192 B
  __shared__ __bf16 spk2S[B_ * HID_];                       //  6400 B
  __shared__ float bhhS[G3_], fcWS[HID_ * OUT_], thrSS[HID_], fcbS[OUT_];

  const int tid  = threadIdx.x;
  const int lane = tid & 31;
  const int wave = tid >> 5;          // 0..7
  const int col  = lane & 15;

  for (int i = tid; i < 2 * 4 * 32 * 16; i += 256) swizMem[i] = (__bf16)0.f;
  for (int i = tid; i < G3_; i += 256) bhhS[i] = bhh[i];
  for (int i = tid; i < HID_ * OUT_; i += 256) fcWS[i] = fcW[i];
  for (int i = tid; i < HID_; i += 256) thrSS[i] = thr_sgru[i];
  if (tid < OUT_) fcbS[tid] = fcb[tid];

  // Whh B-fragments: tile = wave + 8*i -> (mt, nt); held in VGPRs.
  v16bf whhB[5][2];
  int mt_[5], nt_[5];
  #pragma unroll
  for (int i = 0; i < 5; ++i) {
    const int tile = wave + 8 * i;
    mt_[i] = tile / 10;  nt_[i] = tile % 10;
    #pragma unroll
    for (int kc = 0; kc < 2; ++kc)
      #pragma unroll
      for (int h = 0; h < 16; ++h) {
        const int k = kc * 32 + (h & 7) + 8 * (lane >> 4) + ((h >= 8) ? 16 : 0);
        const int n = nt_[i] * 16 + col;
        float v = 0.f;
        if (k < HID_ && n < G3_) v = Whh[k * G3_ + n];
        whhB[i][kc][h] = (__bf16)v;
      }
  }

  float memreg[13];
  int memSlot[13];
  #pragma unroll
  for (int it = 0; it < 13; ++it) {
    memreg[it] = 0.f;
    const int e = tid + it * 256;
    if (e < B_ * HID_) {
      const int b = e / HID_, j = e % HID_;
      const int kc = j >> 5, kl = j & 31;
      const int lnhi = (kl >> 3) & 1;
      const int h = (kl & 7) + ((kl & 16) ? 8 : 0);
      memSlot[it] = ((kc * 4 + (b >> 4)) * 32 + (b & 15) + 16 * lnhi) * 16 + h;
    } else memSlot[it] = 0;
  }
  float ml = 0.f;
  const float beta = beta_p[0];
  const float thrL = thrL_p[0];

  __syncthreads();

  for (int t = 0; t < T_; ++t) {
    // ---- phase A: hg = mem_s @ Whh + bhh (WMMA) ----
    #pragma unroll
    for (int i = 0; i < 5; ++i) {
      v8f acc;
      #pragma unroll
      for (int r = 0; r < 8; ++r) acc[r] = 0.f;
      #pragma unroll
      for (int kc = 0; kc < 2; ++kc) {
        const v16bf af = *reinterpret_cast<const v16bf*>(
            &swizMem[((kc * 4 + mt_[i]) * 32 + lane) * 16]);
        acc = __builtin_amdgcn_wmma_f32_16x16x32_bf16(
            false, af, false, whhB[i][kc], (short)0, acc, false, false);
      }
      const int n = nt_[i] * 16 + col;
      #pragma unroll
      for (int r = 0; r < 8; ++r) {
        const int rw = mt_[i] * 16 + r + ((lane & 16) ? 8 : 0);
        if (n < G3_) hgS[rw * G3_ + n] = (__bf16)(acc[r] + bhhS[n]);
      }
    }
    __syncthreads();

    // ---- phase B: GRU gates, spike, reset-by-subtraction ----
    const float* xg = xp + (size_t)t * B_ * G3_;
    #pragma unroll
    for (int it = 0; it < 13; ++it) {
      const int e = tid + it * 256;
      if (e < B_ * HID_) {
        const int b = e / HID_, j = e % HID_;
        const float hz = (float)hgS[b * G3_ + j];
        const float hr = (float)hgS[b * G3_ + HID_ + j];
        const float hn = (float)hgS[b * G3_ + 2 * HID_ + j];
        const float z = 1.f / (1.f + __expf(-(xg[b * G3_ + j] + hz)));
        const float r = 1.f / (1.f + __expf(-(xg[b * G3_ + HID_ + j] + hr)));
        const float n = tanhf(xg[b * G3_ + 2 * HID_ + j] + r * hn);
        float m = (1.f - z) * n + z * memreg[it];
        const float th = thrSS[j];
        const float s  = (m - th) > 0.f ? 1.f : 0.f;
        m -= s * th;
        memreg[it] = m;
        spk2S[e] = (__bf16)s;
        swizMem[memSlot[it]] = (__bf16)m;   // direct write in fragment order
      }
    }
    __syncthreads();

    // ---- phase C: fc + leaky integrate-and-fire ----
    if (tid < B_ * OUT_) {
      const int b = tid / OUT_, o = tid % OUT_;
      float s = fcbS[o];
      #pragma unroll
      for (int j = 0; j < HID_; ++j)
        s += (float)spk2S[b * HID_ + j] * fcWS[j * OUT_ + o];
      ml = beta * ml + s;
      const float sp = (ml - thrL) > 0.f ? 1.f : 0.f;
      ml -= sp * thrL;
      out[(size_t)t * (B_ * OUT_) + tid] = sp;                          // spk_rec
      out[(size_t)T_ * B_ * OUT_ + (size_t)t * (B_ * OUT_) + tid] = ml; // mem_rec
    }
    __syncthreads();
  }
}

extern "C" void kernel_launch(void* const* d_in, const int* in_sizes, int n_in,
                              void* d_out, int out_size, void* d_ws, size_t ws_size,
                              hipStream_t stream) {
  const float* x    = (const float*)d_in[0];
  const float* Wih  = (const float*)d_in[1];
  const float* Whh  = (const float*)d_in[2];
  const float* bih  = (const float*)d_in[3];
  const float* bhh  = (const float*)d_in[4];
  const float* thrS = (const float*)d_in[5];
  const float* fcW  = (const float*)d_in[6];
  const float* fcb  = (const float*)d_in[7];
  const float* beta = (const float*)d_in[8];
  const float* thrL = (const float*)d_in[9];
  float* out = (float*)d_out;

  __bf16* swizW = (__bf16*)d_ws;
  float*  xp    = (float*)((char*)d_ws + XP_OFFSET_BYTES);  // ~10.9 MB total

  const int nsw = NCH_ * 32 * 160;
  swizzle_wih_kernel<<<dim3((nsw + 255) / 256), dim3(256), 0, stream>>>(Wih, swizW);
  xproj_kernel<<<dim3(13 * 16), dim3(256), 0, stream>>>(x, swizW, bih, xp);
  sgru_scan_kernel<<<dim3(1), dim3(256), 0, stream>>>(xp, Whh, bhh, thrS,
                                                      fcW, fcb, beta, thrL, out);
}
